// GraphAttentionHead_41120016892257
// MI455X (gfx1250) — matrix operand order repaired and checked by reading, not compiled, not run on hardware
//
#include <hip/hip_runtime.h>
#include <hip/hip_bf16.h>

typedef __attribute__((ext_vector_type(16))) _Float16 v16h;
typedef __attribute__((ext_vector_type(8)))  float    v8f;

#define B_     8
#define N_     2048
#define FIN_   128
#define FOUT_  64
#define ALPHA_ 0.2f
#define NEGINF_ -9000000000000000.0f

// ---------------------------------------------------------------------------
// Kernel 0: pack adj (int32 0/1, [N,N]) into a bitmask, 1 bit per entry.
// ---------------------------------------------------------------------------
__global__ __launch_bounds__(256) void gat_pack_adj(const int* __restrict__ adj,
                                                    unsigned* __restrict__ bits) {
  int t = blockIdx.x * blockDim.x + threadIdx.x;       // element index in [0, N*N)
  unsigned lane = threadIdx.x & 31u;
  unsigned b = (adj[t] > 0) ? (1u << lane) : 0u;
  b |= __shfl_xor(b, 1, 32);
  b |= __shfl_xor(b, 2, 32);
  b |= __shfl_xor(b, 4, 32);
  b |= __shfl_xor(b, 8, 32);
  b |= __shfl_xor(b, 16, 32);
  if (lane == 0) bits[t >> 5] = b;
}

// ---------------------------------------------------------------------------
// Kernel 1: h = x @ W (per-wave 16x64 tile via v_wmma_f32_16x16x32_f16),
// plus e1 = h.a1, e2 = h.a2, plus hT (f16, feature-major [B][64][2048]).
//
// WMMA f16 operand layouts (wave32, ISA 7.12.2):
//   A (16x32):  lane (g=lane/16, m=lane%16) holds row M=m,
//               half i -> K = i + 8g + (i>=8 ? 8 : 0)
//   B (32x16):  lane (g,m) holds col N=m, half i -> K = i + 8g + (i>=8 ? 8 : 0)
//   C/D (16x16 f32): float r of v8f -> (M = r + 8g, N = m)
// ---------------------------------------------------------------------------
__global__ __launch_bounds__(128) void gat_proj(const float* __restrict__ x,
                                                const float* __restrict__ W,
                                                const float* __restrict__ a,
                                                _Float16* __restrict__ hT,
                                                float* __restrict__ e1,
                                                float* __restrict__ e2) {
  const int wave = (blockIdx.x * blockDim.x + threadIdx.x) >> 5;
  const int lane = threadIdx.x & 31;
  const int g = lane >> 4;
  const int m = lane & 15;
  const int row0 = wave << 4;                 // tile base row in flattened [B*N]
  const int b    = row0 >> 11;                // row0 / 2048
  const int n0   = row0 & (N_ - 1);           // row within batch

  v8f acc[4] = {};                            // 4 col-tiles of 16 -> 64 features

#pragma unroll
  for (int k0 = 0; k0 < FIN_; k0 += 32) {
    const float* xr = x + (size_t)(row0 + m) * FIN_ + k0 + 8 * g;
    float4 q0 = *(const float4*)(xr);         // K = k0 + 8g + 0..3
    float4 q1 = *(const float4*)(xr + 4);     // K = k0 + 8g + 4..7
    float4 q2 = *(const float4*)(xr + 16);    // K = k0 + 16 + 8g + 0..3
    float4 q3 = *(const float4*)(xr + 20);    // K = k0 + 16 + 8g + 4..7
    v16h A;
    A[0]=(_Float16)q0.x;  A[1]=(_Float16)q0.y;  A[2]=(_Float16)q0.z;  A[3]=(_Float16)q0.w;
    A[4]=(_Float16)q1.x;  A[5]=(_Float16)q1.y;  A[6]=(_Float16)q1.z;  A[7]=(_Float16)q1.w;
    A[8]=(_Float16)q2.x;  A[9]=(_Float16)q2.y;  A[10]=(_Float16)q2.z; A[11]=(_Float16)q2.w;
    A[12]=(_Float16)q3.x; A[13]=(_Float16)q3.y; A[14]=(_Float16)q3.z; A[15]=(_Float16)q3.w;

#pragma unroll
    for (int nt = 0; nt < 4; ++nt) {
      const float* wp = W + (size_t)(k0 + 8 * g) * FOUT_ + nt * 16 + m;
      v16h Bv;
#pragma unroll
      for (int i = 0; i < 16; ++i) Bv[i] = (_Float16)wp[(i + ((i >= 8) ? 8 : 0)) * FOUT_];
      acc[nt] = __builtin_amdgcn_wmma_f32_16x16x32_f16(
          false, A, false, Bv, (short)0, acc[nt], false, false);
    }
  }

  // e1/e2: per-row dot with a1/a2 from C-layout accumulators
  float p1[8], p2[8];
#pragma unroll
  for (int r = 0; r < 8; ++r) { p1[r] = 0.f; p2[r] = 0.f; }
#pragma unroll
  for (int nt = 0; nt < 4; ++nt) {
    float a1v = a[nt * 16 + m];
    float a2v = a[FOUT_ + nt * 16 + m];
#pragma unroll
    for (int r = 0; r < 8; ++r) {
      p1[r] += acc[nt][r] * a1v;
      p2[r] += acc[nt][r] * a2v;
    }
  }
#pragma unroll
  for (int off = 1; off < 16; off <<= 1) {
#pragma unroll
    for (int r = 0; r < 8; ++r) {
      p1[r] += __shfl_xor(p1[r], off, 32);
      p2[r] += __shfl_xor(p2[r], off, 32);
    }
  }
  if (m == 0) {
#pragma unroll
    for (int r = 0; r < 8; ++r) {
      e1[row0 + r + 8 * g] = p1[r];
      e2[row0 + r + 8 * g] = p2[r];
    }
  }

  // hT (f16, feature-major): hT[(b*64 + f)*2048 + n]
#pragma unroll
  for (int nt = 0; nt < 4; ++nt) {
#pragma unroll
    for (int r = 0; r < 8; ++r) {
      hT[(size_t)(b * FOUT_ + nt * 16 + m) * N_ + (n0 + r + 8 * g)] =
          (_Float16)acc[nt][r];
    }
  }
}

// ---------------------------------------------------------------------------
// Kernel 1.5: e2max[b] = max_j e2[b, j]  (one wave per batch).
// With lrelu monotonic, lrelu(e1[i] + e2max[b]) upper-bounds every row max,
// letting the attention kernel run a single-pass (non-rescaling) softmax.
// ---------------------------------------------------------------------------
__global__ __launch_bounds__(256) void gat_e2max(const float* __restrict__ e2,
                                                 float* __restrict__ e2max) {
  const int wave = threadIdx.x >> 5;          // = batch (8 waves in 1 block)
  const int lane = threadIdx.x & 31;
  const float* p = e2 + (size_t)wave * N_;
  float mx = -3.0e38f;
  for (int k = lane; k < N_; k += 32) mx = fmaxf(mx, p[k]);
#pragma unroll
  for (int off = 16; off; off >>= 1) mx = fmaxf(mx, __shfl_xor(mx, off, 32));
  if (lane == 0) e2max[wave] = mx;
}

// ---------------------------------------------------------------------------
// Kernel 2: fused masked-softmax attention.
// Block = 4 waves = 4 consecutive 16-row tiles of ONE batch. The shared
// h[j0:j0+32][0:64] B-tile (4 KB f16) is staged into LDS with
// global_load_async_to_lds_b128 (ASYNCcnt), double-buffered, so L2 traffic
// drops 4x vs per-wave loads. Single-pass softmax with precomputed row bound.
// ---------------------------------------------------------------------------
__global__ __launch_bounds__(128) void gat_attn(const _Float16* __restrict__ hT,
                                                const float* __restrict__ e1,
                                                const float* __restrict__ e2,
                                                const float* __restrict__ e2maxp,
                                                const unsigned* __restrict__ bits,
                                                float* __restrict__ out) {
  __shared__ _Float16 tile[2][FOUT_ * 32];    // [buf][f*32 + j], 8 KB double buffer

  const int tid  = threadIdx.x;
  const int wave = tid >> 5;
  const int lane = tid & 31;
  const int g = lane >> 4;
  const int m = lane & 15;
  const int b  = blockIdx.x >> 5;             // 32 blocks per batch
  const int i0 = (((blockIdx.x & 31) << 2) | wave) << 4;

  const float e1row = e1[b * N_ + i0 + m];
  const float marg  = e1row + e2maxp[b];
  const float Mrow  = (marg > 0.f) ? marg : ALPHA_ * marg;  // >= true row max
  const unsigned* abrow = bits + (size_t)(i0 + m) * (N_ / 32);
  const float* e2b = e2 + (size_t)b * N_;

  // cooperative staging: thread -> (feature lf = tid/2, j-half lh = tid&1),
  // 32 bytes each = two async b128s (inst offset applies to LDS and global).
  const int lf = tid >> 1;
  const int lh = tid & 1;
  const unsigned long long gbase =
      (unsigned long long)(uintptr_t)(hT + (size_t)b * FOUT_ * N_); // uniform
  const unsigned goff_thread = (unsigned)((lf * N_ + lh * 16) * sizeof(_Float16));
  // LDS byte offset; 'tile' is the only __shared__ object -> base offset 0.
  const unsigned lds_thread  = (unsigned)((lf * 32 + lh * 16) * sizeof(_Float16));

#define ISSUE_CHUNK(J0, BUF) do {                                              \
    unsigned _go = goff_thread + (unsigned)((J0) * sizeof(_Float16));          \
    unsigned _lo = lds_thread +                                                \
                   (unsigned)((BUF) * FOUT_ * 32 * sizeof(_Float16));          \
    asm volatile("global_load_async_to_lds_b128 %0, %1, %2 offset:0"           \
                 :: "v"(_lo), "v"(_go), "s"(gbase) : "memory");                \
    asm volatile("global_load_async_to_lds_b128 %0, %1, %2 offset:16"          \
                 :: "v"(_lo), "v"(_go), "s"(gbase) : "memory");                \
  } while (0)

  v8f acc[4] = {};
  float run_sum = 0.0f;

  ISSUE_CHUNK(0, 0);

  for (int c = 0; c < N_ / 32; ++c) {
    const int j0  = c * 32;
    const int buf = c & 1;
    if (c + 1 < N_ / 32) {
      ISSUE_CHUNK(j0 + 32, buf ^ 1);
      asm volatile("s_wait_asynccnt 0x2" ::: "memory"); // current buf complete
    } else {
      asm volatile("s_wait_asynccnt 0x0" ::: "memory");
    }
    __syncthreads();                          // cross-wave visibility of buf

    unsigned word = abrow[j0 >> 5];           // adj bits for cols j0..j0+31
    const float* ep = e2b + j0 + 16 * g;
    float4 q0 = *(const float4*)(ep);
    float4 q1 = *(const float4*)(ep + 4);
    float4 q2 = *(const float4*)(ep + 8);
    float4 q3 = *(const float4*)(ep + 12);
    float v[16] = {q0.x, q0.y, q0.z, q0.w, q1.x, q1.y, q1.z, q1.w,
                   q2.x, q2.y, q2.z, q2.w, q3.x, q3.y, q3.z, q3.w};

    v16h P;
#pragma unroll
    for (int i = 0; i < 16; ++i) {
      const int K = i + 16 * g;                         // col offset in chunk
      float ev = e1row + v[i];
      ev = (ev > 0.0f) ? ev : ALPHA_ * ev;              // leaky_relu
      float p = ((word >> K) & 1u) ? __expf(ev - Mrow) : 0.0f; // mask -> exact 0
      run_sum += p;
      P[i] = (_Float16)p;
    }

    // acc += P(16x32) @ h[j0:j0+32][0:64], B operand from LDS (contiguous 32B)
#pragma unroll
    for (int nt = 0; nt < 4; ++nt) {
      const v16h Bv = *(const v16h*)&tile[buf][(nt * 16 + m) * 32 + 16 * g];
      acc[nt] = __builtin_amdgcn_wmma_f32_16x16x32_f16(
          false, P, false, Bv, (short)0, acc[nt], false, false);
    }
    __syncthreads();                          // all waves done with buf before reuse
  }
#undef ISSUE_CHUNK

  // full row denominators: combine the two j-halves of each row, then divide
  run_sum += __shfl_xor(run_sum, 16, 32);
  const float rinv = 1.0f / run_sum;
#pragma unroll
  for (int r = 0; r < 8; ++r) {
    float s = __shfl(rinv, r + 8 * g, 32);
#pragma unroll
    for (int nt = 0; nt < 4; ++nt) {
      out[(size_t)(b * N_ + i0 + r + 8 * g) * FOUT_ + nt * 16 + m] =
          acc[nt][r] * s;
    }
  }
}

// ---------------------------------------------------------------------------
extern "C" void kernel_launch(void* const* d_in, const int* in_sizes, int n_in,
                              void* d_out, int out_size, void* d_ws, size_t ws_size,
                              hipStream_t stream) {
  const float* x   = (const float*)d_in[0];   // [8,2048,128] f32
  const int*   adj = (const int*)  d_in[1];   // [2048,2048] i32
  const float* W   = (const float*)d_in[2];   // [128,64] f32
  const float* a   = (const float*)d_in[3];   // [128,1] f32
  float* out = (float*)d_out;                 // [8,2048,64] f32

  // workspace: hT f16 (2 MB) | e1 (64 KB) | e2 (64 KB) | bitmask (512 KB) | e2max
  char* ws = (char*)d_ws;
  _Float16* hT = (_Float16*)ws;
  float* e1 = (float*)(ws + (size_t)B_ * FOUT_ * N_ * sizeof(_Float16));
  float* e2 = e1 + B_ * N_;
  unsigned* bits = (unsigned*)(e2 + B_ * N_);
  float* e2max = (float*)(bits + (N_ / 32) * N_);

  // 0) pack adjacency bits
  gat_pack_adj<<<(N_ * N_) / 256, 256, 0, stream>>>(adj, bits);

  // 1) projection: 1024 waves (16 rows each) -> 256 blocks of 128
  gat_proj<<<(B_ * N_ / 16) * 32 / 128, 128, 0, stream>>>(x, W, a, hT, e1, e2);

  // 1.5) per-batch e2 max (8 waves, one block)
  gat_e2max<<<1, 256, 0, stream>>>(e2, e2max);

  // 2) fused attention: 256 blocks x 4 waves, each wave one 16-row tile
  gat_attn<<<(B_ * (N_ / 16)) / 4, 128, 0, stream>>>(hT, e1, e2, e2max, bits, out);
}
